// PharmRecDynamicsGVP_17102559773370
// MI455X (gfx1250) — matrix operand order repaired and checked by reading, not compiled, not run on hardware
//
#include <hip/hip_runtime.h>
#include <math.h>

// ---------------------------------------------------------------------------
// PharmRec GVP dynamics forward for MI455X (gfx1250, wave32, WMMA bf16).
//
// Strategy: the dominant cost is per-edge GVP stacks: GEMMs of shape
// (16-edge tile) x (K=146/144 pad 160) x (N=128) -> bf16 WMMA with f32 acc.
// One wave (32 lanes) owns a 16-row tile; everything (scalar feats, vector
// channels, intermediate activations) stays in LDS across the 3 GVP layers.
// Weights are pre-packed into WMMA-fragment order so each lane's B fragment
// is one contiguous 32B load. Segment sums use global f32 atomics (L2).
// This round: fast v_rcp_f32 sigmoid (avoid precise-division sequence).
// ---------------------------------------------------------------------------

typedef __bf16 bf16_t;
typedef __attribute__((ext_vector_type(16))) __bf16 v16bf;
typedef __attribute__((ext_vector_type(8)))  float  v8f;

#define NFC   512        // pharm nodes
#define NPC   9600       // prot nodes
#define SDIM  128
#define VDIM  16
#define MAXV  17         // max vector channels (layer-1 msg input = V+1)
#define AK    160        // padded K for all big GEMMs (multiple of 32)
#define NKT   (AK/32)    // 5 K-tiles
#define EPPC  (NPC*8)    // pp edge count (== fp/pf edge count)

// Fast sigmoid: v_exp_f32 + v_rcp_f32 (1 ulp), no div_scale/fixup sequence.
__device__ __forceinline__ float sigmoid_f(float x) {
  return __builtin_amdgcn_rcpf(1.f + __expf(-x));
}
__device__ __forceinline__ float silu_f(float x) { return x * sigmoid_f(x); }

struct GvpWArg {
  const bf16_t* Ws;   // packed fragment order, nkt=5, nnt=sout/16
  const bf16_t* Wg;   // packed fragment order, nkt=sout/32, nnt=1 (vout padded to 16)
  const float*  Wh;   // [vin][dh] f32
  const float*  Wu;   // [dh][vout] f32
  const float*  bs;   // [sout]
  const float*  bg;   // [vout]
  int vin, dh, vout, sin, sout, gate_sig;
};

// A fragment: 16x32 bf16 tile from LDS row-major A[16][AK].
// Lane layout (ISA 7.12.2): rows = lane&15; lanes<16 hold K [0..7]+[16..23],
// lanes>=16 hold K [8..15]+[24..31] of the 32-wide K tile.
__device__ __forceinline__ v16bf lds_load_a_frag(const bf16_t* A, int kt, int lane) {
  const bf16_t* p = A + (lane & 15) * AK + kt * 32 + ((lane & 16) ? 8 : 0);
  v16bf a;
#pragma unroll
  for (int i = 0; i < 8; ++i) { a[i] = p[i]; a[8 + i] = p[16 + i]; }
  return a;
}

// B fragment: weights pre-packed so lane reads 16 contiguous bf16 (32B).
__device__ __forceinline__ v16bf g_load_b_frag(const bf16_t* B, int nkt, int kt, int nt, int lane) {
  const bf16_t* p = B + (((size_t)(nt * nkt + kt) * 32) + (size_t)lane) * 16;
  v16bf b;
#pragma unroll
  for (int i = 0; i < 16; ++i) b[i] = p[i];
  return b;
}

// One GVP layer on a 16-row tile held in LDS.
//  Ain : [16][AK] bf16, cols [0,sin) hold s; this writes sh into [sin,sin+dh)
//  Aout: [16][AK] bf16, receives silu(s@Ws+bs) in cols [0,sout)
//  vbuf: [16*MAXV][3] f32 vector channels (in: vin rows, out: vout rows)
__device__ void gvp_apply(int lane, const GvpWArg w,
                          bf16_t* Ain, bf16_t* Aout,
                          float* vbuf, float* vh, float* gate)
{
  const int m    = lane & 15;
  const int half = lane >> 4;

  // ---- Vh = v @ Wh (VALU, 2 lanes/row); sh = ||Vh|| into A; zero K pad ----
  {
    const int hq = (w.dh + 1) >> 1;
    const int h0 = half * hq;
    int h1 = h0 + hq; if (h1 > w.dh) h1 = w.dh;
    for (int h = h0; h < h1; ++h) {
      float a0 = 0.f, a1 = 0.f, a2 = 0.f;
      for (int vv = 0; vv < w.vin; ++vv) {
        const float wv = w.Wh[vv * w.dh + h];
        const float* vp_ = vbuf + (m * MAXV + vv) * 3;
        a0 += vp_[0] * wv; a1 += vp_[1] * wv; a2 += vp_[2] * wv;
      }
      float* hp_ = vh + (m * MAXV + h) * 3;
      hp_[0] = a0; hp_[1] = a1; hp_[2] = a2;
      Ain[m * AK + w.sin + h] = (bf16_t)sqrtf(a0*a0 + a1*a1 + a2*a2 + 1e-8f);
    }
    for (int c = w.sin + w.dh + half; c < AK; c += 2)
      Ain[m * AK + c] = (bf16_t)0.f;
  }
  __syncthreads();

  // ---- s_out = silu(A @ Ws + bs) : WMMA bf16, 5 K-tiles x (sout/16) N-tiles
  {
    const int col = lane & 15;
    const int rb  = (lane & 16) ? 8 : 0;
    const int ntiles = w.sout >> 4;
    for (int nt = 0; nt < ntiles; ++nt) {
      v8f acc = {0.f,0.f,0.f,0.f,0.f,0.f,0.f,0.f};
#pragma unroll
      for (int kt = 0; kt < NKT; ++kt) {
        v16bf af = lds_load_a_frag(Ain, kt, lane);
        v16bf bfrag = g_load_b_frag(w.Ws, NKT, kt, nt, lane);
        acc = __builtin_amdgcn_wmma_f32_16x16x32_bf16(false, af, false, bfrag,
                                                      (short)0, acc, false, false);
      }
      const int n = nt * 16 + col;
      const float bsn = w.bs[n];
#pragma unroll
      for (int j = 0; j < 8; ++j)
        Aout[(rb + j) * AK + n] = (bf16_t)silu_f(acc[j] + bsn);
    }
  }
  __syncthreads();

  // ---- gate = act(s_out @ Wg + bg) : WMMA, N padded to 16 ----
  {
    const int u  = lane & 15;
    const int rb = (lane & 16) ? 8 : 0;
    v8f acc = {0.f,0.f,0.f,0.f,0.f,0.f,0.f,0.f};
    const int nkt = w.sout >> 5;
    for (int kt = 0; kt < nkt; ++kt) {
      v16bf af = lds_load_a_frag(Aout, kt, lane);
      v16bf bfrag = g_load_b_frag(w.Wg, nkt, kt, 0, lane);
      acc = __builtin_amdgcn_wmma_f32_16x16x32_bf16(false, af, false, bfrag,
                                                    (short)0, acc, false, false);
    }
    const float bgu = (u < w.vout) ? w.bg[u] : 0.f;
#pragma unroll
    for (int j = 0; j < 8; ++j) {
      float x = acc[j] + bgu;
      gate[(rb + j) * 16 + u] = w.gate_sig ? sigmoid_f(x) : x;
    }
  }
  __syncthreads();

  // ---- v_out = gate * (Vh @ Wu), overwrite vbuf ----
  {
    const int uq = (w.vout + 1) >> 1;
    const int u0 = half * uq;
    int u1 = u0 + uq; if (u1 > w.vout) u1 = w.vout;
    for (int u = u0; u < u1; ++u) {
      float a0 = 0.f, a1 = 0.f, a2 = 0.f;
      for (int h = 0; h < w.dh; ++h) {
        const float wu = w.Wu[h * w.vout + u];
        const float* hp_ = vh + (m * MAXV + h) * 3;
        a0 += hp_[0] * wu; a1 += hp_[1] * wu; a2 += hp_[2] * wu;
      }
      const float g = gate[m * 16 + u];
      float* vp_ = vbuf + (m * MAXV + u) * 3;
      vp_[0] = g * a0; vp_[1] = g * a1; vp_[2] = g * a2;
    }
  }
  __syncthreads();
}

// ---------------------------------------------------------------------------
// Edge-message kernel: 16 edges per wave, fused gather + 3 GVP layers +
// atomic segment-sum scatter. src_idx/dst_idx==nullptr means idx = e/8.
// ---------------------------------------------------------------------------
__global__ __launch_bounds__(32) void edge_msg_kernel(
    const float* __restrict__ h_src, const float* __restrict__ v_src,
    const float* __restrict__ x_src, const float* __restrict__ x_dst,
    const int* __restrict__ src_idx, const int* __restrict__ dst_idx,
    GvpWArg g0, GvpWArg g1, GvpWArg g2,
    float* __restrict__ agg_s, float* __restrict__ agg_v)
{
  __shared__ bf16_t A0[16 * AK];
  __shared__ bf16_t A1[16 * AK];
  __shared__ float  vbuf[16 * MAXV * 3];
  __shared__ float  vh[16 * MAXV * 3];
  __shared__ float  gate[256];
  __shared__ int    sdst[16];

  const int lane = threadIdx.x;
  const int m    = lane & 15;
  const int half = lane >> 4;
  const int e    = blockIdx.x * 16 + m;

  __builtin_prefetch(g0.Ws, 0, 0);   // global_prefetch_b8 on weight stream

  const int si = src_idx ? src_idx[e] : (e >> 3);
  const int di = dst_idx ? dst_idx[e] : (e >> 3);
  if (half == 0) sdst[m] = di;

  const float dx = x_src[si*3+0] - x_dst[di*3+0];
  const float dy = x_src[si*3+1] - x_dst[di*3+1];
  const float dz = x_src[si*3+2] - x_dst[di*3+2];
  const float d  = sqrtf(dx*dx + dy*dy + dz*dz + 1e-8f);

  for (int c = 0; c < 64; ++c)
    A0[m*AK + half*64 + c] = (bf16_t)h_src[si*SDIM + half*64 + c];
  if (half == 0) {
    A0[m*AK + SDIM] = (bf16_t)d;          // s = [h_src, d], sin = 129
    const float inv = __builtin_amdgcn_rcpf(d);
    float* r0 = vbuf + (m*MAXV + 0) * 3;  // channel 0 = xd/d
    r0[0] = dx*inv; r0[1] = dy*inv; r0[2] = dz*inv;
  }
  for (int u = half*8; u < half*8 + 8; ++u) {
    float* vp_ = vbuf + (m*MAXV + 1 + u) * 3;
    vp_[0] = v_src[(si*VDIM + u)*3 + 0];
    vp_[1] = v_src[(si*VDIM + u)*3 + 1];
    vp_[2] = v_src[(si*VDIM + u)*3 + 2];
  }
  __syncthreads();

  gvp_apply(lane, g0, A0, A1, vbuf, vh, gate);
  gvp_apply(lane, g1, A1, A0, vbuf, vh, gate);
  gvp_apply(lane, g2, A0, A1, vbuf, vh, gate);

  const int dd = sdst[m];
  for (int c = 0; c < 64; ++c)
    atomicAdd(&agg_s[dd*SDIM + half*64 + c], (float)A1[m*AK + half*64 + c]);
  for (int u = half*8; u < half*8 + 8; ++u) {
    const float* vp_ = vbuf + (m*MAXV + u) * 3;
    atomicAdd(&agg_v[(dd*VDIM + u)*3 + 0], vp_[0]);
    atomicAdd(&agg_v[(dd*VDIM + u)*3 + 1], vp_[1]);
    atomicAdd(&agg_v[(dd*VDIM + u)*3 + 2], vp_[2]);
  }
}

// ---------------------------------------------------------------------------
// Node update: h = LN(h+agg); v += agg; (s,v) through 2-GVP stack;
// h = LN(h+su); v += vu. 16 nodes per wave.
// ---------------------------------------------------------------------------
__global__ __launch_bounds__(32) void node_update_kernel(
    float* __restrict__ h, float* __restrict__ v,
    const float* __restrict__ agg_s, const float* __restrict__ agg_v,
    const float* __restrict__ g1, const float* __restrict__ b1,
    const float* __restrict__ g2, const float* __restrict__ b2,
    GvpWArg w0, GvpWArg w1)
{
  __shared__ bf16_t A0[16 * AK];
  __shared__ bf16_t A1[16 * AK];
  __shared__ float  vbuf[16 * MAXV * 3];
  __shared__ float  vh[16 * MAXV * 3];
  __shared__ float  gate[256];
  __shared__ float  hcur[16 * SDIM];
  __shared__ float  redS[32], redQ[32];

  const int lane = threadIdx.x;
  const int m    = lane & 15;
  const int half = lane >> 4;
  const int node = blockIdx.x * 16 + m;

  // ---- LN1(h + agg_s) ----
  float ps = 0.f, pq = 0.f;
  for (int c = 0; c < 64; ++c) {
    const int cc = half*64 + c;
    float x = h[node*SDIM + cc] + agg_s[node*SDIM + cc];
    hcur[m*SDIM + cc] = x; ps += x; pq += x*x;
  }
  redS[lane] = ps; redQ[lane] = pq;
  __syncthreads();
  {
    const float mu  = (redS[m] + redS[m+16]) * (1.f/128.f);
    const float var = (redQ[m] + redQ[m+16]) * (1.f/128.f) - mu*mu;
    const float rs  = rsqrtf(var + 1e-5f);
    for (int c = 0; c < 64; ++c) {
      const int cc = half*64 + c;
      float x = (hcur[m*SDIM + cc] - mu) * rs * g1[cc] + b1[cc];
      hcur[m*SDIM + cc] = x;
      A0[m*AK + cc] = (bf16_t)x;
    }
  }
  for (int u = half*8; u < half*8 + 8; ++u) {
    float* vp_ = vbuf + (m*MAXV + u) * 3;
    vp_[0] = v[(node*VDIM+u)*3+0] + agg_v[(node*VDIM+u)*3+0];
    vp_[1] = v[(node*VDIM+u)*3+1] + agg_v[(node*VDIM+u)*3+1];
    vp_[2] = v[(node*VDIM+u)*3+2] + agg_v[(node*VDIM+u)*3+2];
  }
  __syncthreads();

  gvp_apply(lane, w0, A0, A1, vbuf, vh, gate);
  gvp_apply(lane, w1, A1, A0, vbuf, vh, gate);   // su in A0, vu in vbuf

  // ---- LN2(hcur + su) ----
  ps = 0.f; pq = 0.f;
  for (int c = 0; c < 64; ++c) {
    const int cc = half*64 + c;
    float x = hcur[m*SDIM + cc] + (float)A0[m*AK + cc];
    hcur[m*SDIM + cc] = x; ps += x; pq += x*x;
  }
  __syncthreads();
  redS[lane] = ps; redQ[lane] = pq;
  __syncthreads();
  {
    const float mu  = (redS[m] + redS[m+16]) * (1.f/128.f);
    const float var = (redQ[m] + redQ[m+16]) * (1.f/128.f) - mu*mu;
    const float rs  = rsqrtf(var + 1e-5f);
    for (int c = 0; c < 64; ++c) {
      const int cc = half*64 + c;
      h[node*SDIM + cc] = (hcur[m*SDIM + cc] - mu) * rs * g2[cc] + b2[cc];
    }
  }
  for (int u = half*8; u < half*8 + 8; ++u) {
    const float* vp_ = vbuf + (m*MAXV + u) * 3;
    v[(node*VDIM+u)*3+0] += agg_v[(node*VDIM+u)*3+0] + vp_[0];
    v[(node*VDIM+u)*3+1] += agg_v[(node*VDIM+u)*3+1] + vp_[1];
    v[(node*VDIM+u)*3+2] += agg_v[(node*VDIM+u)*3+2] + vp_[2];
  }
}

// ---------------------------------------------------------------------------
// Noise head: 3 GVPs (last with identity gate, sout=64, vout=1), then
// eps_h = s @ Wout(64,8) + bout ; eps_x = v[:,0,:].
// ---------------------------------------------------------------------------
__global__ __launch_bounds__(32) void noise_kernel(
    const float* __restrict__ hf, const float* __restrict__ vf,
    GvpWArg w0, GvpWArg w1, GvpWArg w2,
    const float* __restrict__ Wout, const float* __restrict__ bout,
    float* __restrict__ out)
{
  __shared__ bf16_t A0[16 * AK];
  __shared__ bf16_t A1[16 * AK];
  __shared__ float  vbuf[16 * MAXV * 3];
  __shared__ float  vh[16 * MAXV * 3];
  __shared__ float  gate[256];

  const int lane = threadIdx.x;
  const int m    = lane & 15;
  const int half = lane >> 4;
  const int node = blockIdx.x * 16 + m;

  for (int c = 0; c < 64; ++c)
    A0[m*AK + half*64 + c] = (bf16_t)hf[node*SDIM + half*64 + c];
  for (int u = half*8; u < half*8 + 8; ++u) {
    float* vp_ = vbuf + (m*MAXV + u) * 3;
    vp_[0] = vf[(node*VDIM+u)*3+0];
    vp_[1] = vf[(node*VDIM+u)*3+1];
    vp_[2] = vf[(node*VDIM+u)*3+2];
  }
  __syncthreads();

  gvp_apply(lane, w0, A0, A1, vbuf, vh, gate);
  gvp_apply(lane, w1, A1, A0, vbuf, vh, gate);
  gvp_apply(lane, w2, A0, A1, vbuf, vh, gate);  // s(64) in A1, eps_x in vbuf ch0

  for (int o = half*4; o < half*4 + 4; ++o) {
    float acc = bout[o];
    for (int k = 0; k < 64; ++k)
      acc += (float)A1[m*AK + k] * Wout[k*8 + o];
    out[node*8 + o] = acc;
  }
  if (half == 0) {
    const float* vp_ = vbuf + (m*MAXV + 0) * 3;
    out[NFC*8 + node*3 + 0] = vp_[0];
    out[NFC*8 + node*3 + 1] = vp_[1];
    out[NFC*8 + node*3 + 2] = vp_[2];
  }
}

// ---------------------------------------------------------------------------
// Encoders: h = LN(silu([h0, t] @ W + b)); one node per block (128 threads).
// ---------------------------------------------------------------------------
__global__ __launch_bounds__(128) void encoder_kernel(
    const float* __restrict__ h0, int nin,
    const float* __restrict__ t, const int* __restrict__ batch,
    const float* __restrict__ W, const float* __restrict__ b,
    const float* __restrict__ g, const float* __restrict__ be,
    float* __restrict__ hout)
{
  __shared__ float rs[128], rq[128];
  const int node = blockIdx.x, c = threadIdx.x;
  float acc = b[c];
  for (int i = 0; i < nin; ++i) acc += h0[node*nin + i] * W[i*SDIM + c];
  acc += t[batch[node]] * W[nin*SDIM + c];
  const float x = silu_f(acc);
  rs[c] = x; rq[c] = x*x;
  __syncthreads();
  for (int s2 = 64; s2 > 0; s2 >>= 1) {
    if (c < s2) { rs[c] += rs[c+s2]; rq[c] += rq[c+s2]; }
    __syncthreads();
  }
  const float mu  = rs[0] * (1.f/128.f);
  const float var = rq[0] * (1.f/128.f) - mu*mu;
  hout[node*SDIM + c] = (x - mu) * rsqrtf(var + 1e-5f) * g[c] + be[c];
}

// ---------------------------------------------------------------------------
// kNN: 16 same-batch candidates per node, select 8 nearest (ties -> low idx).
// ---------------------------------------------------------------------------
__global__ __launch_bounds__(64) void knn_ff_kernel(const float* __restrict__ x,
                                                    int* __restrict__ nbr)
{
  const int i = blockIdx.x * 64 + threadIdx.x;
  if (i >= NFC) return;
  const int base = (i >> 4) << 4;
  const float x0 = x[i*3], x1 = x[i*3+1], x2 = x[i*3+2];
  float d2[16];
  for (int j = 0; j < 16; ++j) {
    const int jj = base + j;
    const float a = x[jj*3]-x0, b = x[jj*3+1]-x1, c = x[jj*3+2]-x2;
    d2[j] = (jj == i) ? 1e30f : a*a + b*b + c*c;
  }
  for (int k = 0; k < 8; ++k) {
    int bj = 0; float bd = 1e38f;
    for (int j = 0; j < 16; ++j) if (d2[j] < bd) { bd = d2[j]; bj = j; }
    nbr[i*8 + k] = base + bj;
    d2[bj] = 1e38f;
  }
}

__global__ __launch_bounds__(256) void knn_pf_kernel(const float* __restrict__ qx,
                                                     const float* __restrict__ px,
                                                     int* __restrict__ nbr)
{
  const int p = blockIdx.x * 256 + threadIdx.x;
  if (p >= NPC) return;
  const int base = (p / 300) * 16;
  const float y0 = qx[p*3], y1 = qx[p*3+1], y2 = qx[p*3+2];
  float d2[16];
  for (int j = 0; j < 16; ++j) {
    const int jj = base + j;
    const float a = y0-px[jj*3], b = y1-px[jj*3+1], c = y2-px[jj*3+2];
    d2[j] = a*a + b*b + c*c;
  }
  for (int k = 0; k < 8; ++k) {
    int bj = 0; float bd = 1e38f;
    for (int j = 0; j < 16; ++j) if (d2[j] < bd) { bd = d2[j]; bj = j; }
    nbr[p*8 + k] = base + bj;
    d2[bj] = 1e38f;
  }
}

// ---------------------------------------------------------------------------
// Weight pre-pack: f32 (K,N) row-major -> bf16 in WMMA B-fragment order.
// dst index = ((nt*nkt + kt)*32 + lane)*16 + i ;  k = kt*32 + 8*lane[4] + (i&7) + 16*i[3],
// n = nt*16 + (lane&15). OOB -> 0 (K pad / N pad).
// ---------------------------------------------------------------------------
__global__ __launch_bounds__(256) void pack_bfrag_kernel(
    const float* __restrict__ src, bf16_t* __restrict__ dst,
    int K, int N, int nkt, int nnt)
{
  const int tot = nkt * nnt * 512;
  const int idx = blockIdx.x * 256 + threadIdx.x;
  if (idx >= tot) return;
  const int i    = idx & 15;
  const int lane = (idx >> 4) & 31;
  const int tile = idx >> 9;
  const int kt = tile % nkt;
  const int nt = tile / nkt;
  const int k = kt * 32 + ((lane & 16) ? 8 : 0) + (i & 7) + ((i & 8) ? 16 : 0);
  const int n = nt * 16 + (lane & 15);
  dst[idx] = (bf16_t)((k < K && n < N) ? src[k * N + n] : 0.f);
}

// ---------------------------------------------------------------------------
// Host orchestration.
// Leaf order assumption: top-level dict in setup_inputs() insertion order;
// the 'params' pytree flattened jax-style (dicts in sorted-key order, lists
// in order). params keys sorted: convs, noise, pharm_enc, prot_enc.
// Per conv (sorted: ln1, ln2, msg, upd): ln1{pharm(g,b),prot(g,b)}=4,
// ln2=4, msg{ff,fp,pf,pp} x 3 gvps x {Wg,Wh,Ws,Wu,bg,bs}=72,
// upd{pharm,prot} x 2 gvps x 6 = 24  -> 104 leaves/conv.
// noise: Wout, bout, gvps(3x6)=20. enc: W,b,be,g = 4 each. Total 444 + 8 ins.
// ---------------------------------------------------------------------------
extern "C" void kernel_launch(void* const* d_in, const int* in_sizes, int n_in,
                              void* d_out, int out_size, void* d_ws, size_t ws_size,
                              hipStream_t stream)
{
  (void)in_sizes; (void)n_in; (void)out_size; (void)ws_size;

  auto F = [&](int i){ return (const float*)d_in[i]; };
  auto I = [&](int i){ return (const int*)d_in[i]; };

  enum { L_WG = 0, L_WH = 1, L_WS = 2, L_WU = 3, L_BG = 4, L_BS = 5 };
  auto msg_leaf = [](int c, int t, int l, int w){ return c*104 + 8  + t*18 + l*6 + w; };
  auto upd_leaf = [](int c, int t, int l, int w){ return c*104 + 80 + t*12 + l*6 + w; };
  auto noi_leaf = [](int l, int w){ return 418 + l*6 + w; };
  const int NOISE_WOUT = 416, NOISE_BOUT = 417;
  const int PE_W = 436, PE_B = 437, PE_BE = 438, PE_G = 439;
  const int QE_W = 440, QE_B = 441, QE_BE = 442, QE_G = 443;
  const int IN_PH0 = 444, IN_QH0 = 445, IN_PX = 446, IN_QX = 447;
  const int IN_T = 448, IN_PB = 449, IN_QB = 450, IN_PP = 451;

  // ---- workspace arena ----
  char* wsb = (char*)d_ws;
  size_t off = 0;
  auto alloc = [&](size_t bytes)->char* {
    char* p = wsb + off;
    off = (off + bytes + 255) & ~(size_t)255;
    return p;
  };
  float* hf    = (float*)alloc((size_t)NFC*SDIM*4);
  float* hp    = (float*)alloc((size_t)NPC*SDIM*4);
  float* vf    = (float*)alloc((size_t)NFC*VDIM*3*4);
  float* vp    = (float*)alloc((size_t)NPC*VDIM*3*4);
  float* aggsF = (float*)alloc((size_t)NFC*SDIM*4);
  float* aggvF = (float*)alloc((size_t)NFC*VDIM*3*4);
  float* aggsP = (float*)alloc((size_t)NPC*SDIM*4);
  float* aggvP = (float*)alloc((size_t)NPC*VDIM*3*4);
  int*   ff_nbr = (int*)alloc((size_t)NFC*8*4);
  int*   pf_nbr = (int*)alloc((size_t)NPC*8*4);

  const size_t WS128 = (size_t)AK*128*sizeof(bf16_t);
  const size_t WS64  = (size_t)AK*64 *sizeof(bf16_t);
  const size_t WG128 = (size_t)128*16*sizeof(bf16_t);
  const size_t WG64  = (size_t)64 *16*sizeof(bf16_t);

  bf16_t *msgWs[4][4][3], *msgWg[4][4][3];
  bf16_t *updWs[4][2][2], *updWg[4][2][2];
  bf16_t *noiWs[3], *noiWg[3];
  for (int c = 0; c < 4; ++c)
    for (int t = 0; t < 4; ++t)
      for (int l = 0; l < 3; ++l) {
        msgWs[c][t][l] = (bf16_t*)alloc(WS128);
        msgWg[c][t][l] = (bf16_t*)alloc(WG128);
      }
  for (int c = 0; c < 4; ++c)
    for (int t = 0; t < 2; ++t)
      for (int l = 0; l < 2; ++l) {
        updWs[c][t][l] = (bf16_t*)alloc(WS128);
        updWg[c][t][l] = (bf16_t*)alloc(WG128);
      }
  for (int l = 0; l < 3; ++l) {
    noiWs[l] = (bf16_t*)alloc(l == 2 ? WS64 : WS128);
    noiWg[l] = (bf16_t*)alloc(l == 2 ? WG64 : WG128);
  }

  auto pack = [&](const float* src, bf16_t* dst, int K, int N, int nkt, int nnt) {
    const int tot = nkt * nnt * 512;
    pack_bfrag_kernel<<<(tot + 255)/256, 256, 0, stream>>>(src, dst, K, N, nkt, nnt);
  };
  for (int c = 0; c < 4; ++c)
    for (int t = 0; t < 4; ++t)
      for (int l = 0; l < 3; ++l) {
        pack(F(msg_leaf(c,t,l,L_WS)), msgWs[c][t][l], l == 0 ? 146 : 144, 128, NKT, 8);
        pack(F(msg_leaf(c,t,l,L_WG)), msgWg[c][t][l], 128, 16, 4, 1);
      }
  for (int c = 0; c < 4; ++c)
    for (int t = 0; t < 2; ++t)
      for (int l = 0; l < 2; ++l) {
        pack(F(upd_leaf(c,t,l,L_WS)), updWs[c][t][l], 144, 128, NKT, 8);
        pack(F(upd_leaf(c,t,l,L_WG)), updWg[c][t][l], 128, 16, 4, 1);
      }
  for (int l = 0; l < 3; ++l) {
    if (l < 2) {
      pack(F(noi_leaf(l,L_WS)), noiWs[l], 144, 128, NKT, 8);
      pack(F(noi_leaf(l,L_WG)), noiWg[l], 128, 16, 4, 1);
    } else {
      pack(F(noi_leaf(l,L_WS)), noiWs[l], 144, 64, NKT, 4);
      pack(F(noi_leaf(l,L_WG)), noiWg[l], 64, 1, 2, 1);
    }
  }

  auto mkarg = [&](const bf16_t* Ws, const bf16_t* Wg, int iWh, int iWu, int iBS, int iBG,
                   int vin, int dh, int vout, int sin, int sout, int sig) {
    GvpWArg a;
    a.Ws = Ws; a.Wg = Wg;
    a.Wh = F(iWh); a.Wu = F(iWu); a.bs = F(iBS); a.bg = F(iBG);
    a.vin = vin; a.dh = dh; a.vout = vout; a.sin = sin; a.sout = sout; a.gate_sig = sig;
    return a;
  };
  auto msg_arg = [&](int c, int t, int l) {
    return mkarg(msgWs[c][t][l], msgWg[c][t][l],
                 msg_leaf(c,t,l,L_WH), msg_leaf(c,t,l,L_WU),
                 msg_leaf(c,t,l,L_BS), msg_leaf(c,t,l,L_BG),
                 l == 0 ? 17 : 16, l == 0 ? 17 : 16, 16, l == 0 ? 129 : 128, 128, 1);
  };
  auto upd_arg = [&](int c, int t, int l) {
    return mkarg(updWs[c][t][l], updWg[c][t][l],
                 upd_leaf(c,t,l,L_WH), upd_leaf(c,t,l,L_WU),
                 upd_leaf(c,t,l,L_BS), upd_leaf(c,t,l,L_BG),
                 16, 16, 16, 128, 128, 1);
  };
  auto noi_arg = [&](int l) {
    return mkarg(noiWs[l], noiWg[l],
                 noi_leaf(l,L_WH), noi_leaf(l,L_WU),
                 noi_leaf(l,L_BS), noi_leaf(l,L_BG),
                 16, 16, l == 2 ? 1 : 16, 128, l == 2 ? 64 : 128, l == 2 ? 0 : 1);
  };

  const float* px = F(IN_PX);
  const float* qx = F(IN_QX);
  const int* pp_src = I(IN_PP);
  const int* pp_dst = I(IN_PP) + EPPC;

  // ---- init state ----
  hipMemsetAsync(vf, 0, (size_t)NFC*VDIM*3*4, stream);
  hipMemsetAsync(vp, 0, (size_t)NPC*VDIM*3*4, stream);
  encoder_kernel<<<NFC, 128, 0, stream>>>(F(IN_PH0), 8,  F(IN_T), I(IN_PB),
                                          F(PE_W), F(PE_B), F(PE_G), F(PE_BE), hf);
  encoder_kernel<<<NPC, 128, 0, stream>>>(F(IN_QH0), 20, F(IN_T), I(IN_QB),
                                          F(QE_W), F(QE_B), F(QE_G), F(QE_BE), hp);
  knn_ff_kernel<<<(NFC + 63)/64, 64, 0, stream>>>(px, ff_nbr);
  knn_pf_kernel<<<(NPC + 255)/256, 256, 0, stream>>>(qx, px, pf_nbr);

  // ---- convs (msg dict sorted order: ff=0, fp=1, pf=2, pp=3) ----
  for (int c = 0; c < 4; ++c) {
    hipMemsetAsync(aggsF, 0, (size_t)NFC*SDIM*4, stream);
    hipMemsetAsync(aggvF, 0, (size_t)NFC*VDIM*3*4, stream);
    hipMemsetAsync(aggsP, 0, (size_t)NPC*SDIM*4, stream);
    hipMemsetAsync(aggvP, 0, (size_t)NPC*VDIM*3*4, stream);

    // ff: pharm->pharm, src = knn neighbor, dst = e/8
    edge_msg_kernel<<<(NFC*8)/16, 32, 0, stream>>>(
        hf, vf, px, px, ff_nbr, nullptr,
        msg_arg(c,0,0), msg_arg(c,0,1), msg_arg(c,0,2), aggsF, aggvF);
    // pf: prot->pharm, src = e/8 (prot), dst = knn pharm
    edge_msg_kernel<<<EPPC/16, 32, 0, stream>>>(
        hp, vp, qx, px, nullptr, pf_nbr,
        msg_arg(c,2,0), msg_arg(c,2,1), msg_arg(c,2,2), aggsF, aggvF);
    // fp: pharm->prot, src = knn pharm, dst = e/8 (prot)
    edge_msg_kernel<<<EPPC/16, 32, 0, stream>>>(
        hf, vf, px, qx, pf_nbr, nullptr,
        msg_arg(c,1,0), msg_arg(c,1,1), msg_arg(c,1,2), aggsP, aggvP);
    // pp: prot->prot from given edge list
    edge_msg_kernel<<<EPPC/16, 32, 0, stream>>>(
        hp, vp, qx, qx, pp_src, pp_dst,
        msg_arg(c,3,0), msg_arg(c,3,1), msg_arg(c,3,2), aggsP, aggvP);

    node_update_kernel<<<NFC/16, 32, 0, stream>>>(
        hf, vf, aggsF, aggvF,
        F(c*104 + 0), F(c*104 + 1), F(c*104 + 4), F(c*104 + 5),
        upd_arg(c,0,0), upd_arg(c,0,1));
    node_update_kernel<<<NPC/16, 32, 0, stream>>>(
        hp, vp, aggsP, aggvP,
        F(c*104 + 2), F(c*104 + 3), F(c*104 + 6), F(c*104 + 7),
        upd_arg(c,1,0), upd_arg(c,1,1));
  }

  // ---- noise head -> d_out ([NF*8] eps_h, then [NF*3] eps_x) ----
  noise_kernel<<<NFC/16, 32, 0, stream>>>(
      hf, vf, noi_arg(0), noi_arg(1), noi_arg(2),
      F(NOISE_WOUT), F(NOISE_BOUT), (float*)d_out);
}
